// SupConLoss_16621523436339
// MI455X (gfx1250) — compile-verified, matrix operand order checked by reading, not run on hardware
//
#include <hip/hip_runtime.h>
#include <hip/hip_bf16.h>

// ---------------------------------------------------------------------------
// SupConLoss fused pipeline for MI455X (gfx1250, wave32, WMMA).
//
//   B = 4096 rows, D = 256 feature dim, C = 10 classes.
//   S = feats_norm @ feats_norm^T  (4096x4096) is computed strip-by-strip:
//   each workgroup owns 16 rows, computes the 16x4096 strip with
//   v_wmma_f32_16x16x32_bf16 into LDS (256KB), then fuses ALL reductions
//   (loss1 logsumexp + positive sums, loss2 per-class masked max/sum-exp)
//   out of LDS. The 64MB Gram matrix never reaches HBM.
// ---------------------------------------------------------------------------

#define BB   4096
#define DD   256
#define NCLS 10
#define T1I  10.0f   // 1/TEMP  (TEMP  = 0.1)
#define T2I  20.0f   // 1/TEMP2 (TEMP2 = 0.05)

typedef __attribute__((ext_vector_type(16))) __bf16 v16bf;
typedef __attribute__((ext_vector_type(8)))  float  v8f;

// ---- wave32 reductions ----------------------------------------------------
__device__ __forceinline__ float wred_max(float x) {
#pragma unroll
  for (int m = 16; m > 0; m >>= 1) x = fmaxf(x, __shfl_xor(x, m, 32));
  return x;
}
__device__ __forceinline__ float wred_sum(float x) {
#pragma unroll
  for (int m = 16; m > 0; m >>= 1) x += __shfl_xor(x, m, 32);
  return x;
}

// ---- A/B fragment loader (ISA 16-bit A-matrix 16x32 layout) ---------------
// lane l: row = row0 + (l%16); K chunks [8*hi, 8*hi+8) and [16+8*hi, ...+8)
__device__ __forceinline__ v16bf load_frag(const __bf16* __restrict__ base,
                                           int row0, int kb, int lane) {
  const int m  = lane & 15;
  const int hi = (lane >> 4) & 1;
  const __bf16* p = base + (size_t)(row0 + m) * DD + kb + 8 * hi;
  v16bf f;
#pragma unroll
  for (int e = 0; e < 8; ++e) {
    f[e]     = p[e];        // 16B contiguous -> global_load_b128
    f[e + 8] = p[e + 16];   // second 16B chunk
  }
  return f;
}

// ---------------------------------------------------------------------------
// Kernel 0: zero global accumulators.
// ws layout (dwords): [0]=ceSum [1]=l1Sum [2..11]=classSums [12..21]=hist(int)
// ---------------------------------------------------------------------------
__global__ void k_init(float* acc, int* hist) {
  int t = threadIdx.x;
  if (t < 12) acc[t] = 0.0f;
  if (t >= 12 && t < 22) hist[t - 12] = 0;
}

// ---------------------------------------------------------------------------
// Kernel 1: L2-normalize rows, emit bf16 feats (one wave per row).
// ---------------------------------------------------------------------------
__global__ void k_normalize(const float* __restrict__ feats,
                            __bf16* __restrict__ out) {
  const int lane = threadIdx.x & 31;
  const int row  = blockIdx.x * (blockDim.x >> 5) + (threadIdx.x >> 5);
  if (row >= BB) return;
  const float* f = feats + (size_t)row * DD;
  float ss = 0.0f;
#pragma unroll
  for (int j = lane; j < DD; j += 32) { float x = f[j]; ss += x * x; }
  ss = wred_sum(ss);
  const float scale = 1.0f / fmaxf(sqrtf(ss), 1e-12f);
#pragma unroll
  for (int j = lane; j < DD; j += 32)
    out[(size_t)row * DD + j] = (__bf16)(f[j] * scale);
}

// ---------------------------------------------------------------------------
// Kernel 2: cross-entropy on predicts (one thread per row, C=10).
// ---------------------------------------------------------------------------
__global__ void k_ce(const float* __restrict__ predicts,
                     const int* __restrict__ targets, float* acc) {
  const int i = blockIdx.x * blockDim.x + threadIdx.x;
  if (i >= BB) return;
  const float* p = predicts + (size_t)i * NCLS;
  float m = -3.0e38f;
#pragma unroll
  for (int c = 0; c < NCLS; ++c) m = fmaxf(m, p[c]);
  float s = 0.0f;
#pragma unroll
  for (int c = 0; c < NCLS; ++c) s += __expf(p[c] - m);
  const int t = targets[i];
  const float logp = p[t] - m - __logf(s);
  atomicAdd(&acc[0], -logp);
}

// ---------------------------------------------------------------------------
// Kernel 3: label histogram.
// ---------------------------------------------------------------------------
__global__ void k_hist(const int* __restrict__ targets, int* hist) {
  const int i = blockIdx.x * blockDim.x + threadIdx.x;
  if (i < BB) atomicAdd(&hist[targets[i]], 1);
}

// ---------------------------------------------------------------------------
// Kernel 4: main fused strip kernel. 256 blocks x 512 threads (16 waves).
// Dynamic LDS: 16*4096 f32 strip (256KB) + 4096 labels (16KB) + partials.
// ---------------------------------------------------------------------------
__global__ void __launch_bounds__(512, 1)
k_supcon(const __bf16* __restrict__ feats, const int* __restrict__ labels,
         float* acc) {
  extern __shared__ char smem[];
  float* sS   = (float*)smem;                                // [16][4096]
  int*   sLab = (int*)(smem + (size_t)16 * BB * 4);          // [4096]
  float* sP   = (float*)(smem + (size_t)16 * BB * 4 + BB * 4); // [0..9]=cls, [10]=l1

  const int rb   = blockIdx.x * 16;
  const int tid  = threadIdx.x;
  const int lane = tid & 31;
  const int wave = tid >> 5;

  for (int j = tid; j < BB; j += 512) sLab[j] = labels[j];
  if (tid < 11) sP[tid] = 0.0f;
  __syncthreads();

  // ---- Phase 1: WMMA GEMM, 16x4096 strip into LDS ----
  // Hoist the block's 8 A-fragments (K = 0..255 in steps of 32).
  v16bf aF[8];
#pragma unroll
  for (int ks = 0; ks < 8; ++ks) aF[ks] = load_frag(feats, rb, ks * 32, lane);

  const int n  = lane & 15;
  const int hi = lane >> 4;
  for (int t = wave; t < BB / 16; t += 16) {
    const int cb = t * 16;
    v8f c = {};
#pragma unroll
    for (int ks = 0; ks < 8; ++ks) {
      v16bf b = load_frag(feats, cb, ks * 32, lane);
      c = __builtin_amdgcn_wmma_f32_16x16x32_bf16(
          /*neg_a=*/false, aF[ks], /*neg_b=*/false, b,
          /*c_mod=*/(short)0, c, /*reuse_a=*/false, /*reuse_b=*/false);
    }
    // C layout: lane -> (N = lane%16, M = v + 8*(lane/16))
#pragma unroll
    for (int v = 0; v < 8; ++v)
      sS[(size_t)(v + 8 * hi) * BB + cb + n] = c[v];
  }
  __syncthreads();

  // ---- Phase 2: fused reductions; wave w owns strip row w ----
  {
    const int r     = wave;
    const int gr    = rb + r;
    const int myLab = sLab[gr];
    const float* row = sS + (size_t)r * BB;

    // loss1: rowmax of s1 (diag zeroed), positive sum/count
    float m1 = -3.0e38f, posS = 0.0f, posC = 0.0f;
    for (int j = lane; j < BB; j += 32) {
      const float s1 = (j == gr) ? 0.0f : row[j] * T1I;
      m1 = fmaxf(m1, s1);
      if (sLab[j] == myLab && j != gr) { posS += s1; posC += 1.0f; }
    }
    m1 = wred_max(m1); posS = wred_sum(posS); posC = wred_sum(posC);

    float den = 0.0f;
    for (int j = lane; j < BB; j += 32) {
      const float s1 = (j == gr) ? 0.0f : row[j] * T1I;
      den += __expf(s1 - m1);
    }
    den = wred_sum(den);
    if (lane == 0 && posC > 0.5f) {
      const float pl = posS / posC - m1 - __logf(den + 1e-12f);
      atomicAdd(&sP[10], pl);   // ds_add_f32
    }

    // loss2: for each class c != myLab, restricted rowmax + sum-exp.
    // val[j] = (lab[j]!=myLab) ? S2 : 0, columns restricted to lab[j]!=c.
    for (int c2 = 0; c2 < NCLS; ++c2) {
      float m = -3.0e38f;
      for (int j = lane; j < BB; j += 32) {
        const int lj = sLab[j];
        if (lj == c2) continue;
        const float v = (lj != myLab) ? row[j] * T2I : 0.0f;
        m = fmaxf(m, v);
      }
      m = wred_max(m);
      float rs = 0.0f;
      for (int j = lane; j < BB; j += 32) {
        const int lj = sLab[j];
        if (lj == c2) continue;
        const float v = (lj != myLab) ? row[j] * T2I : 0.0f;
        rs += __expf(v - m);
      }
      rs = wred_sum(rs);
      if (lane == 0 && myLab != c2) atomicAdd(&sP[c2], rs);  // ds_add_f32
    }
  }
  __syncthreads();

  if (tid < NCLS) atomicAdd(&acc[2 + tid], sP[tid]);
  if (tid == NCLS) atomicAdd(&acc[1], sP[10]);
}

// ---------------------------------------------------------------------------
// Kernel 5: combine scalars -> out[0].
// ---------------------------------------------------------------------------
__global__ void k_final(const float* acc, const int* hist, float* out) {
  if (threadIdx.x != 0 || blockIdx.x != 0) return;
  const float Bf = (float)BB;
  const float ce = acc[0] / Bf;           // mean CE
  const float cl = -acc[1] / Bf;          // -mean(pos_logits)

  float h[NCLS], sumsq = 0.0f;
  for (int c = 0; c < NCLS; ++c) { h[c] = (float)hist[c]; sumsq += h[c] * h[c]; }
  float ncnt[NCLS]; bool allzero = true;
  for (int c = 0; c < NCLS; ++c) {
    const float Nc = Bf - h[c];
    ncnt[c] = Nc * Nc - (sumsq - h[c] * h[c]);  // diff-label pairs in neg(c)
    if (ncnt[c] != 0.0f) allzero = false;
  }
  float tls = 0.0f;
  for (int c = 0; c < NCLS; ++c) {
    if (h[c] >= 2.0f) {                    // rows with mask_sum > 0
      float x = acc[2 + c];
      if (!allzero) x = x / ncnt[c];
      tls += h[c] * __logf(x + 1e-12f);
    }
  }
  const float tl = tls / Bf;               // -mean(pos_logits) for loss2
  // (1-ALPHA)*ce + ALPHA*cl + 0.5*ALPHA*tl, ALPHA = 0.5
  out[0] = 0.5f * ce + 0.5f * cl + 0.25f * tl;
}

// ---------------------------------------------------------------------------
extern "C" void kernel_launch(void* const* d_in, const int* in_sizes, int n_in,
                              void* d_out, int out_size, void* d_ws, size_t ws_size,
                              hipStream_t stream) {
  const float* cls_feats = (const float*)d_in[0];   // [4096,256] f32
  const float* predicts  = (const float*)d_in[1];   // [4096,10]  f32
  const int*   targets   = (const int*)d_in[2];     // [4096]     i32
  float* out = (float*)d_out;

  float*  acc   = (float*)d_ws;                     // 12 f32 accumulators
  int*    hist  = (int*)d_ws + 12;                  // 10 int counts
  __bf16* featsB = (__bf16*)((char*)d_ws + 512);    // 4096*256 bf16 (2MB)

  k_init<<<1, 32, 0, stream>>>(acc, hist);
  k_normalize<<<BB / 8, 256, 0, stream>>>(cls_feats, featsB);
  k_ce<<<BB / 256, 256, 0, stream>>>(predicts, targets, acc);
  k_hist<<<BB / 256, 256, 0, stream>>>(targets, hist);

  const size_t shbytes = (size_t)16 * BB * 4   // S strip  (256KB)
                       + (size_t)BB * 4        // labels   (16KB)
                       + 16 * 4;               // partials
  k_supcon<<<BB / 16, 512, shbytes, stream>>>(featsB, targets, acc);
  k_final<<<1, 32, 0, stream>>>(acc, hist, out);
}
